// Loreclayer_ft_67353677136188
// MI455X (gfx1250) — matrix operand order repaired and checked by reading, not compiled, not run on hardware
//
#include <hip/hip_runtime.h>
#include <hip/hip_bf16.h>
#include <stdint.h>

typedef __attribute__((ext_vector_type(16))) __bf16 v16bf;
typedef __attribute__((ext_vector_type(8)))  __bf16 v8bf;
typedef __attribute__((ext_vector_type(8)))  float  v8f;

constexpr int   B_   = 8;
constexpr int   S_   = 2048;
constexpr int   DIN  = 5120;
constexpr int   DOUT = 5120;
constexpr int   R_   = 64;
constexpr int   CTRH = 256;
constexpr int   CTRF = 128;
constexpr float LNEPS   = 1e-5f;
constexpr float SCALE_  = 128.0f / 64.0f;   // ALPHA / R
constexpr int   ROWS    = B_ * S_;          // 16384

// ---- workspace byte offsets (all 256B aligned) ----
constexpr size_t OFF_SA   = 0;                       //  8*64*4      = 2 KB
constexpr size_t OFF_CF   = 4096;                    //  8*128*4     = 4 KB
constexpr size_t OFF_SB   = 12288;                   //  8*5120*4    = 160 KB
constexpr size_t OFF_WAPK = 12288 + 163840;          //  64*5120*2   = 640 KB (bf16 fragments)
constexpr size_t OFF_WBPK = OFF_WAPK + 655360;       //  5120*64*2   = 640 KB
constexpr size_t OFF_A    = OFF_WBPK + 655360;       //  2 slabs * 16384*64*4 = 8 MB (fp32 K-partials)
// total ~9.8 MB

__device__ __forceinline__ void async_g2l_b128(uint32_t lds_byte_off, uint64_t gaddr) {
  // VDST = per-lane LDS byte offset, VADDR = per-lane 64-bit global address (GV mode)
  asm volatile("global_load_async_to_lds_b128 %0, %1, off"
               :: "v"(lds_byte_off), "v"(gaddr) : "memory");
}

__device__ __forceinline__ v8f wmma_bf16(v16bf a, v16bf b, v8f c) {
  return __builtin_amdgcn_wmma_f32_16x16x32_bf16(false, a, false, b, (short)0, c, false, false);
}

// ---------------------------------------------------------------------------
// prep1: LayerNorm(ctr_hidden) -> ctr_final -> sA        (single block, 256 thr)
// ---------------------------------------------------------------------------
__global__ void __launch_bounds__(256) k_prep1(
    const float* __restrict__ ctr, const float* __restrict__ gamma,
    const float* __restrict__ beta, const float* __restrict__ Wc,
    const float* __restrict__ bc,  const float* __restrict__ WAad,
    float* __restrict__ cf_ws, float* __restrict__ sA_ws) {
  __shared__ float h[B_][CTRH];
  __shared__ float cf[B_][CTRF];
  const int tid = threadIdx.x, w = tid >> 5, l = tid & 31;

  // wave w normalizes row w (8 elems/lane, lane-strided for coalescing)
  float v[8];
  float s = 0.f;
  #pragma unroll
  for (int j = 0; j < 8; ++j) { v[j] = ctr[w * CTRH + l + 32 * j]; s += v[j]; }
  #pragma unroll
  for (int o = 16; o >= 1; o >>= 1) s += __shfl_xor(s, o, 32);
  const float mu = s * (1.f / CTRH);
  float vs = 0.f;
  #pragma unroll
  for (int j = 0; j < 8; ++j) { float d = v[j] - mu; vs += d * d; }
  #pragma unroll
  for (int o = 16; o >= 1; o >>= 1) vs += __shfl_xor(vs, o, 32);
  const float rstd = rsqrtf(vs * (1.f / CTRH) + LNEPS);
  #pragma unroll
  for (int j = 0; j < 8; ++j) {
    int e = l + 32 * j;
    h[w][e] = (v[j] - mu) * rstd * gamma[e] + beta[e];
  }
  __syncthreads();

  // ctr_final[b][j] = h[b] . Wc[j] + bc[j]   (1024 outputs)
  #pragma unroll
  for (int q = 0; q < 4; ++q) {
    int idx = tid + q * 256, b = idx >> 7, jj = idx & 127;
    float acc = bc[jj];
    for (int d = 0; d < CTRH; ++d) acc += h[b][d] * Wc[jj * CTRH + d];
    cf[b][jj] = acc;
    cf_ws[idx] = acc;
  }
  __syncthreads();

  // sA[b][r] = cf[b] . WAad[r]   (512 outputs)
  #pragma unroll
  for (int q = 0; q < 2; ++q) {
    int idx = tid + q * 256, b = idx >> 6, r = idx & 63;
    float acc = 0.f;
    for (int d = 0; d < CTRF; ++d) acc += cf[b][d] * WAad[r * CTRF + d];
    sA_ws[idx] = acc;
  }
}

// ---------------------------------------------------------------------------
// prep2: sB[b][o] = (ctr_final[b] . WBad[o]) * SCALING   (40960 outputs)
// ---------------------------------------------------------------------------
__global__ void __launch_bounds__(256) k_prep2(
    const float* __restrict__ cf_ws, const float* __restrict__ WBad,
    float* __restrict__ sB_ws) {
  __shared__ float cf[B_ * CTRF];
  const int tid = threadIdx.x;
  #pragma unroll
  for (int q = 0; q < 4; ++q) cf[tid + q * 256] = cf_ws[tid + q * 256];
  __syncthreads();
  const int idx = blockIdx.x * 256 + tid;           // grid sized exactly 40960
  const int b = idx / DOUT, o = idx % DOUT;
  float acc = 0.f;
  for (int d = 0; d < CTRF; ++d) acc += cf[b * CTRF + d] * WBad[o * CTRF + d];
  sB_ws[idx] = acc * SCALE_;
}

// ---------------------------------------------------------------------------
// repack W_A [64,5120] fp32 -> bf16 B-fragments (tile = kt*4+nt, 512 elems)
// B frag k order used here (A matches): lane l, col n=l&15, K = (l<16 ? 0..15 : 16..31) consecutive
// ---------------------------------------------------------------------------
__global__ void __launch_bounds__(256) k_repackA(
    const float* __restrict__ WA, __bf16* __restrict__ WApk) {
  const int cid = blockIdx.x * 256 + threadIdx.x;   // 20480 lane-chunks
  const int t = cid >> 5, l = cid & 31;
  const int kt = t >> 2, nt = t & 3;
  const int n  = nt * 16 + (l & 15);
  const int k0 = kt * 32 + ((l & 16) ? 16 : 0);
  const float* src = WA + (size_t)n * DIN + k0;     // contiguous 16 floats
  __bf16* dst = WApk + (size_t)t * 512 + l * 16;
  #pragma unroll
  for (int j = 0; j < 16; ++j) dst[j] = (__bf16)src[j];
}

// repack W_B [5120,64] fp32 -> bf16 B-fragments (tile = nt*2+kt)
__global__ void __launch_bounds__(256) k_repackB(
    const float* __restrict__ WB, __bf16* __restrict__ WBpk) {
  const int cid = blockIdx.x * 256 + threadIdx.x;   // 20480 lane-chunks
  const int t = cid >> 5, l = cid & 31;
  const int nt = t >> 1, kt = t & 1;
  const int n  = nt * 16 + (l & 15);
  const int k0 = kt * 32 + ((l & 16) ? 16 : 0);
  const float* src = WB + (size_t)n * R_ + k0;      // contiguous 16 floats
  __bf16* dst = WBpk + (size_t)t * 512 + l * 16;
  #pragma unroll
  for (int j = 0; j < 16; ++j) dst[j] = (__bf16)src[j];
}

// ---------------------------------------------------------------------------
// gemm1: partial_a[slab][row][r] = (x[:, slabK] @ W_A[:, slabK]^T) * sA[b][r]
// grid (256, 2): 256 row-blocks (64 rows = 4 waves) x 2 K-slabs of 2560.
// 5-buffer async LDS pipeline, 3 chunks in flight; 6 async instrs / iter.
// Deterministic: each slab writes its own fp32 partial buffer (no atomics);
// gemm2 sums the two partials.
// ---------------------------------------------------------------------------
__global__ void __launch_bounds__(128) k_gemm1(
    const float* __restrict__ x, const __bf16* __restrict__ WApk,
    const float* __restrict__ sA_ws, float* __restrict__ aPart) {
  constexpr int NB  = 5;
  constexpr int NIT = 2560 / 32;                   // 80 K-steps per slab
  __shared__ float  bufA[NB][64 * 32];             // 8 KB per buffer
  __shared__ __attribute__((aligned(64))) __bf16 bufB[NB][4 * 512];  // 4 KB per buffer

  const int tid = threadIdx.x, w = tid >> 5, l = tid & 31;
  const int rowsBase = blockIdx.x * 64;
  const int waveRow  = rowsBase + w * 16;
  const int slab     = blockIdx.y;                 // 0 or 1
  const int kOff     = slab * 2560;                // slab K origin (cols of x)
  const int ktOff    = kOff / 32;                  // slab origin in 32-k tiles

  auto stage = [&](int kt, int bb) {
    // x tile: 64 rows x 32 k x 4B = 8 KB = 512 x 16B pieces; 4 per thread
    #pragma unroll
    for (int p = 0; p < 4; ++p) {
      int piece = tid * 4 + p;
      int r = piece >> 3, pc = piece & 7;
      uint64_t g = (uint64_t)(uintptr_t)(x + (size_t)(rowsBase + r) * DIN + kOff + kt * 32 + pc * 4);
      uint32_t d = (uint32_t)(uintptr_t)(&bufA[bb][r * 32 + pc * 4]);
      async_g2l_b128(d, g);
    }
    // W_A fragments: 4 n-tiles * 1 KB = 4 KB; 2 x 16B pieces per thread
    #pragma unroll
    for (int p = 0; p < 2; ++p) {
      uint64_t g = (uint64_t)(uintptr_t)(WApk + (size_t)(ktOff + kt) * 2048 + tid * 16 + p * 8);
      uint32_t d = (uint32_t)(uintptr_t)(&bufB[bb][tid * 16 + p * 8]);
      async_g2l_b128(d, g);
    }
  };

  v8f c0 = {}, c1 = {}, c2 = {}, c3 = {};
  #pragma unroll
  for (int kt = 0; kt < 4; ++kt) stage(kt, kt);

  const int mloc  = l & 15;
  const int kbase = (l < 16) ? 0 : 16;             // contiguous k order, matches WApk packing
  int bbC = 0, bbS = 4;                            // rotating buffer indices

  for (int kt = 0; kt < NIT; ++kt) {
    // chunk kt done once <= 3 younger chunks (6 async instrs each) outstanding
    if (kt + 3 < NIT) asm volatile("s_wait_asynccnt 18" ::: "memory");
    else              asm volatile("s_wait_asynccnt 0"  ::: "memory");
    __syncthreads();

    // B fragments first (independent registers -> loads can overlap cvts)
    v16bf b0 = *(const v16bf*)&bufB[bbC][0 * 512 + l * 16];
    v16bf b1 = *(const v16bf*)&bufB[bbC][1 * 512 + l * 16];
    v16bf b2 = *(const v16bf*)&bufB[bbC][2 * 512 + l * 16];
    v16bf b3 = *(const v16bf*)&bufB[bbC][3 * 512 + l * 16];

    // A fragment: rows of x tile, fp32 -> bf16
    const float4* ar4 = (const float4*)&bufA[bbC][(w * 16 + mloc) * 32 + kbase];
    float4 f0 = ar4[0], f1 = ar4[1], f2 = ar4[2], f3 = ar4[3];
    v16bf a;
    a[0]=(__bf16)f0.x; a[1]=(__bf16)f0.y; a[2]=(__bf16)f0.z; a[3]=(__bf16)f0.w;
    a[4]=(__bf16)f1.x; a[5]=(__bf16)f1.y; a[6]=(__bf16)f1.z; a[7]=(__bf16)f1.w;
    a[8]=(__bf16)f2.x; a[9]=(__bf16)f2.y; a[10]=(__bf16)f2.z; a[11]=(__bf16)f2.w;
    a[12]=(__bf16)f3.x; a[13]=(__bf16)f3.y; a[14]=(__bf16)f3.z; a[15]=(__bf16)f3.w;

    c0 = wmma_bf16(a, b0, c0);
    c1 = wmma_bf16(a, b1, c1);
    c2 = wmma_bf16(a, b2, c2);
    c3 = wmma_bf16(a, b3, c3);

    if (kt + 4 < NIT) stage(kt + 4, bbS);
    bbC = (bbC + 1 == NB) ? 0 : bbC + 1;
    bbS = (bbS + 1 == NB) ? 0 : bbS + 1;
  }

  // epilogue: scale by sA[b][r], store fp32 partial (slab-private, no atomics)
  const int b = rowsBase >> 11;                    // 2048 rows per batch
  const int rOff = (l < 16) ? 0 : 8;
  float* aP = aPart + (size_t)slab * ROWS * R_;
  v8f cc[4] = {c0, c1, c2, c3};
  #pragma unroll
  for (int nt = 0; nt < 4; ++nt) {
    float sAv = sA_ws[b * 64 + nt * 16 + mloc];
    #pragma unroll
    for (int i = 0; i < 8; ++i) {
      int row = waveRow + i + rOff;
      aP[(size_t)row * R_ + nt * 16 + mloc] = cc[nt][i] * sAv;
    }
  }
}

// ---------------------------------------------------------------------------
// gemm2: out[row][o] = ((a0+a1) @ W_B^T)[row][o] * sB_scaled[b][o]
// grid (256, 4): 256 row-blocks (64 rows = 4 waves) x 4 N-slices
// (1280 cols each = 10 chunks of 128). 8 async instrs / iter.
// ---------------------------------------------------------------------------
__global__ void __launch_bounds__(128) k_gemm2(
    const float* __restrict__ aPart, const __bf16* __restrict__ WBpk,
    const float* __restrict__ sB_ws, float* __restrict__ out) {
  constexpr int NB  = 5;
  constexpr int NCH = 10;                          // chunks per slice (8 n-tiles each)
  __shared__ __attribute__((aligned(64))) __bf16 buf[NB][8192];   // 16 KB per buffer

  const int tid = threadIdx.x, w = tid >> 5, l = tid & 31;
  const int rowsBase = blockIdx.x * 64;
  const int waveRow  = rowsBase + w * 16;
  const int b = rowsBase >> 11;
  const int chunk0 = blockIdx.y * NCH;

  auto stage = [&](int c, int bb) {
    const __bf16* src = WBpk + (size_t)c * 8192;   // 16 tiles = 16 KB contiguous
    #pragma unroll
    for (int p = 0; p < 8; ++p) {
      uint64_t g = (uint64_t)(uintptr_t)(src + tid * 64 + p * 8);
      uint32_t d = (uint32_t)(uintptr_t)(&buf[bb][tid * 64 + p * 8]);
      async_g2l_b128(d, g);
    }
  };

  // A fragments: sum the two fp32 K-partials, convert to bf16 (deterministic)
  const int mloc = l & 15;
  const int kb   = (l < 16) ? 0 : 16;
  const float* ap0 = aPart + (size_t)(waveRow + mloc) * R_;
  const float* ap1 = ap0 + (size_t)ROWS * R_;
  v16bf a0, a1;
  #pragma unroll
  for (int grp = 0; grp < 4; ++grp) {              // k-groups: kb, kb+8, kb+32, kb+40
    int off = kb + (grp >> 1) * 32 + (grp & 1) * 8;
    float4 p0 = *(const float4*)(ap0 + off);
    float4 p1 = *(const float4*)(ap0 + off + 4);
    float4 q0 = *(const float4*)(ap1 + off);
    float4 q1 = *(const float4*)(ap1 + off + 4);
    __bf16 e0 = (__bf16)(p0.x + q0.x), e1 = (__bf16)(p0.y + q0.y);
    __bf16 e2 = (__bf16)(p0.z + q0.z), e3 = (__bf16)(p0.w + q0.w);
    __bf16 e4 = (__bf16)(p1.x + q1.x), e5 = (__bf16)(p1.y + q1.y);
    __bf16 e6 = (__bf16)(p1.z + q1.z), e7 = (__bf16)(p1.w + q1.w);
    if (grp < 2) {
      int base = grp * 8;
      a0[base+0]=e0; a0[base+1]=e1; a0[base+2]=e2; a0[base+3]=e3;
      a0[base+4]=e4; a0[base+5]=e5; a0[base+6]=e6; a0[base+7]=e7;
    } else {
      int base = (grp - 2) * 8;
      a1[base+0]=e0; a1[base+1]=e1; a1[base+2]=e2; a1[base+3]=e3;
      a1[base+4]=e4; a1[base+5]=e5; a1[base+6]=e6; a1[base+7]=e7;
    }
  }

  #pragma unroll
  for (int i = 0; i < 4; ++i) stage(chunk0 + i, i);

  const int rOff = (l < 16) ? 0 : 8;
  int bbC = 0, bbS = 4;
  for (int ci = 0; ci < NCH; ++ci) {
    if (ci + 3 < NCH) asm volatile("s_wait_asynccnt 24" ::: "memory");
    else              asm volatile("s_wait_asynccnt 0"  ::: "memory");
    __syncthreads();

    #pragma unroll
    for (int ntl = 0; ntl < 8; ++ntl) {
      v16bf b0 = *(const v16bf*)&buf[bbC][(ntl * 2 + 0) * 512 + l * 16];
      v16bf b1 = *(const v16bf*)&buf[bbC][(ntl * 2 + 1) * 512 + l * 16];
      v8f acc = {};
      acc = wmma_bf16(a0, b0, acc);
      acc = wmma_bf16(a1, b1, acc);
      const int o0 = (chunk0 + ci) * 128 + ntl * 16;
      const float sBv = sB_ws[b * DOUT + o0 + mloc];
      #pragma unroll
      for (int i = 0; i < 8; ++i)
        out[(size_t)(waveRow + i + rOff) * DOUT + o0 + mloc] = acc[i] * sBv;
    }

    if (ci + 4 < NCH) stage(chunk0 + ci + 4, bbS);
    bbC = (bbC + 1 == NB) ? 0 : bbC + 1;
    bbS = (bbS + 1 == NB) ? 0 : bbS + 1;
  }
}

// ---------------------------------------------------------------------------
extern "C" void kernel_launch(void* const* d_in, const int* in_sizes, int n_in,
                              void* d_out, int out_size, void* d_ws, size_t ws_size,
                              hipStream_t stream) {
  (void)in_sizes; (void)n_in; (void)out_size; (void)ws_size;
  const float* x    = (const float*)d_in[0];
  const float* ctr  = (const float*)d_in[1];
  const float* gam  = (const float*)d_in[2];
  const float* bet  = (const float*)d_in[3];
  const float* Wc   = (const float*)d_in[4];
  const float* bc   = (const float*)d_in[5];
  const float* WAad = (const float*)d_in[6];
  const float* WBad = (const float*)d_in[7];
  const float* WA   = (const float*)d_in[8];
  const float* WB   = (const float*)d_in[9];

  char* ws = (char*)d_ws;
  float*  sA_ws = (float*)(ws + OFF_SA);
  float*  cf_ws = (float*)(ws + OFF_CF);
  float*  sB_ws = (float*)(ws + OFF_SB);
  __bf16* WApk  = (__bf16*)(ws + OFF_WAPK);
  __bf16* WBpk  = (__bf16*)(ws + OFF_WBPK);
  float*  aPart = (float*)(ws + OFF_A);
  float*  outp  = (float*)d_out;

  k_prep1  <<<1,   256, 0, stream>>>(ctr, gam, bet, Wc, bc, WAad, cf_ws, sA_ws);
  k_prep2  <<<160, 256, 0, stream>>>(cf_ws, WBad, sB_ws);
  k_repackA<<<80,  256, 0, stream>>>(WA, WApk);
  k_repackB<<<80,  256, 0, stream>>>(WB, WBpk);
  k_gemm1  <<<dim3(ROWS / 64, 2),           128, 0, stream>>>(x, WApk, sA_ws, aPart);
  k_gemm2  <<<dim3(ROWS / 64, DOUT / 1280), 128, 0, stream>>>(aPart, WBpk, sB_ws, outp);
}
